// SVTransformerBPETokenV1_63754494542668
// MI455X (gfx1250) — matrix operand order repaired and checked by reading, not compiled, not run on hardware
//
#include <hip/hip_runtime.h>

typedef _Float16 h8   __attribute__((ext_vector_type(8)));
typedef _Float16 h16v __attribute__((ext_vector_type(16)));
typedef float    v8f  __attribute__((ext_vector_type(8)));

// ---------------------------------------------------------------------------
// Problem constants (from reference): V=2048 L=16 H=8 D=64 B=8192, TEMP=8
// ---------------------------------------------------------------------------
#define NV   2048
#define NL   16
#define NH   8
#define ND   64
#define NB   8192
#define NNAMES 2048

// ---------------------------------------------------------------------------
// Kernel 1: cqkv = char_emb @ Wc + bc  ->  f16 tables
//   tabQK[h][v][0..63]  = cq[v,h,:],  tabQK[h][v][64..127] = ck[v,h,:]
//   tabV [h][v][0..63]  = cv[v,h,:]
// ---------------------------------------------------------------------------
__global__ void k_prep_char(const float* __restrict__ emb,
                            const float* __restrict__ Wc,
                            const float* __restrict__ bc,
                            _Float16* __restrict__ tabQK,
                            _Float16* __restrict__ tabV) {
  int idx = blockIdx.x * blockDim.x + threadIdx.x;   // NV*1536 threads
  if (idx >= NV * 1536) return;
  int v = idx / 1536, o = idx % 1536;
  float acc = bc[o];
  const float* er = emb + (size_t)v * ND;
#pragma unroll 8
  for (int k = 0; k < ND; ++k) acc = fmaf(er[k], Wc[(size_t)k * 1536 + o], acc);
  int g = o >> 9, h = (o >> 6) & 7, d = o & 63;
  if (g == 2) tabV[((size_t)h * NV + v) * ND + d] = (_Float16)acc;
  else        tabQK[((size_t)h * NV + v) * 128 + g * 64 + d] = (_Float16)acc;
}

// ---------------------------------------------------------------------------
// Kernel 2: pqkv = pos @ Wp + bp -> posQ/posK/posV[h][l][64] (f16)
// pos computed in-kernel (sinusoidal)
// ---------------------------------------------------------------------------
__global__ void k_prep_pos(const float* __restrict__ Wp,
                           const float* __restrict__ bp,
                           _Float16* __restrict__ posQ,
                           _Float16* __restrict__ posK,
                           _Float16* __restrict__ posV) {
  __shared__ float pm[NL * ND];
  int tid = threadIdx.x;
  for (int i = tid; i < NL * ND; i += blockDim.x) {
    int l = i >> 6, k = i & 63, f = k & 31;
    float inv = __expf(-(float)f * (9.210340371976184f / 32.0f)); // ln(1e4)/freq
    float ph = (float)l * inv;
    pm[i] = (k < 32) ? __cosf(ph) : __sinf(ph);
  }
  __syncthreads();
  for (int j = tid; j < NL * 1536; j += blockDim.x) {
    int l = j / 1536, o = j % 1536;
    float acc = bp[o];
#pragma unroll 8
    for (int k = 0; k < ND; ++k) acc = fmaf(pm[l * ND + k], Wp[(size_t)k * 1536 + o], acc);
    int g = o >> 9, h = (o >> 6) & 7, d = o & 63;
    _Float16* dst = (g == 0) ? posQ : (g == 1) ? posK : posV;
    dst[((size_t)h * NL + l) * ND + d] = (_Float16)acc;
  }
}

// ---------------------------------------------------------------------------
// Kernel 3: per-batch attention + LN + masked pool + MLP.
// One block (128 thr = 4 waves) per batch; each wave owns 2 heads.
// simT = Akey(16x192) x Bqry(192x16) via 6x v_wmma_f32_16x16x32_f16,
// out  = attn(16x32,pad) x V(32x64,pad) via 4x wmma (attn repacked in-register).
// ---------------------------------------------------------------------------
__launch_bounds__(128)
__global__ void k_attn(const _Float16* __restrict__ tabQK,
                       const _Float16* __restrict__ tabV,
                       const _Float16* __restrict__ posQ,
                       const _Float16* __restrict__ posK,
                       const _Float16* __restrict__ posV,
                       const int*      __restrict__ code,
                       const float*    __restrict__ gamma,
                       const float*    __restrict__ beta,
                       const float*    __restrict__ W1,
                       const float*    __restrict__ b1,
                       const float*    __restrict__ W2,
                       const float*    __restrict__ b2,
                       float*          __restrict__ wordx) {
  __shared__ int   cds[NL];
  __shared__ float pooled[NH * ND];

  const int b    = blockIdx.x;
  const int tid  = threadIdx.x;
  const int lane = tid & 31;
  const int wave = tid >> 5;
  if (tid < NL) cds[tid] = code[(size_t)b * NL + tid];
  __syncthreads();

  float cnt = 0.f;
#pragma unroll
  for (int i = 0; i < NL; ++i) cnt += (cds[i] != 0) ? 1.f : 0.f;
  const float invCnt = 1.f / cnt;

  const int col  = lane & 15;   // i (query col of simT) == j (key row) == d%16
  const int half = lane >> 4;
  const int ccol = cds[col];    // token id at position `col`

  float gm[4], bt[4], msk[8];
#pragma unroll
  for (int t = 0; t < 4; ++t) { gm[t] = gamma[t * 16 + col]; bt[t] = beta[t * 16 + col]; }
#pragma unroll
  for (int r = 0; r < 8; ++r) msk[r] = (cds[r + 8 * half] != 0) ? 1.f : 0.f;

  for (int rep = 0; rep < 2; ++rep) {
    const int h = wave * 2 + rep;
    const _Float16* tQK = tabQK + (size_t)h * NV * 128;
    const _Float16* tV  = tabV  + (size_t)h * NV * ND;
    const _Float16* pV  = posV + (size_t)h * NL * ND;
    const _Float16* rowTok = tQK  + (size_t)ccol * 128;          // [cq|ck] of token
    const _Float16* rowPQ  = posQ + ((size_t)h * NL + col) * ND; // pq[col]
    const _Float16* rowPK  = posK + ((size_t)h * NL + col) * ND; // pk[col]

    // query-side row i=col: [cq[ci](0..63) | ck[ci](64..127) | pq[i](128..191)]
    auto ldQry = [&](int c0) -> h8 {
      if (c0 < 128) return *(const h8*)(rowTok + c0);
      return *(const h8*)(rowPQ + (c0 - 128));
    };
    // key-side row j=col: [ck[cj]+pk[j](0..63) | pq[j](64..127) | pk[j](128..191)]
    auto ldKey = [&](int c0) -> h8 {
      if (c0 < 64)  { h8 a = *(const h8*)(rowTok + 64 + c0);
                      h8 p = *(const h8*)(rowPK + c0); return a + p; }
      if (c0 < 128) return *(const h8*)(rowPQ + (c0 - 64));
      return *(const h8*)(rowPK + (c0 - 128));
    };

    // ---- simT = Key x Qry^T : 6 chained K=32 WMMA steps over K=192 ----
    v8f acc = {};
#pragma unroll
    for (int q = 0; q < 6; ++q) {
      union { h16v v; h8 c[2]; } fa, fb;
      int baseA = 32 * q + 8 * half;       // A layout: two 8-chunks
      fa.c[0] = ldKey(baseA);
      fa.c[1] = ldKey(baseA + 16);
      int baseB = 32 * q + 8 * half;       // B layout: two 8-chunks
      fb.c[0] = ldQry(baseB);
      fb.c[1] = ldQry(baseB + 16);
      acc = __builtin_amdgcn_wmma_f32_16x16x32_f16(false, fa.v, false, fb.v,
                                                   (short)0, acc, false, false);
    }

    // ---- mask PAD keys, scale 1/TEMP, softmax over j (rows of simT) ----
    float sv[8];
#pragma unroll
    for (int r = 0; r < 8; ++r) {
      float x = acc[r];
      if (cds[r + 8 * half] == 0) x = -1e9f;
      sv[r] = x * 0.125f;
    }
    float m = sv[0];
#pragma unroll
    for (int r = 1; r < 8; ++r) m = fmaxf(m, sv[r]);
    m = fmaxf(m, __shfl_xor(m, 16, 32));
    float s = 0.f;
#pragma unroll
    for (int r = 0; r < 8; ++r) { sv[r] = __expf(sv[r] - m); s += sv[r]; }
    s += __shfl_xor(s, 16, 32);
    const float invS = 1.f / s;

    // simT C-layout registers == attn A-fragment layout: repack in-register
    union { h16v v; _Float16 e[16]; } fat;
#pragma unroll
    for (int r = 0; r < 8; ++r) fat.e[r] = (_Float16)(sv[r] * invS);
#pragma unroll
    for (int r = 8; r < 16; ++r) fat.e[r] = (_Float16)0.f;

    // ---- out = attn @ (cv[cj]+pv[j]) : 4 N-tiles of 16 ----
    v8f oacc[4];
#pragma unroll
    for (int t = 0; t < 4; ++t) {
      union { h16v v; _Float16 e[16]; } fv;
      {                                    // elems 0-7 hold j=8*half+s; 8-15 = K>=16 (padding)
        const int d = t * 16 + col;
#pragma unroll
        for (int s = 0; s < 8; ++s) {
          const int j = 8 * half + s;
          fv.e[s] = (_Float16)((float)tV[(size_t)cds[j] * ND + d] +
                               (float)pV[(size_t)j * ND + d]);
        }
#pragma unroll
        for (int s = 8; s < 16; ++s) fv.e[s] = (_Float16)0.f;
      }
      v8f z = {};
      oacc[t] = __builtin_amdgcn_wmma_f32_16x16x32_f16(false, fat.v, false, fv.v,
                                                       (short)0, z, false, false);
    }

    // ---- LayerNorm over d (per row i), then masked pool over i ----
    float mu[8], rs[8];
#pragma unroll
    for (int r = 0; r < 8; ++r) {
      float sm = oacc[0][r] + oacc[1][r] + oacc[2][r] + oacc[3][r];
      float sq = oacc[0][r]*oacc[0][r] + oacc[1][r]*oacc[1][r] +
                 oacc[2][r]*oacc[2][r] + oacc[3][r]*oacc[3][r];
#pragma unroll
      for (int mk = 1; mk < 16; mk <<= 1) {
        sm += __shfl_xor(sm, mk, 32);
        sq += __shfl_xor(sq, mk, 32);
      }
      float mean = sm * (1.f / 64.f);
      float var  = sq * (1.f / 64.f) - mean * mean;
      mu[r] = mean;
      rs[r] = rsqrtf(var + 1e-5f);
    }
#pragma unroll
    for (int t = 0; t < 4; ++t) {
      float ps = 0.f;
#pragma unroll
      for (int r = 0; r < 8; ++r) {
        float y = (oacc[t][r] - mu[r]) * rs[r] * gm[t] + bt[t];
        ps += msk[r] * y;
      }
      ps += __shfl_xor(ps, 16, 32);
      if (half == 0) pooled[h * ND + t * 16 + col] = ps * invCnt;
    }
  }
  __syncthreads();

  // ---- tiny MLP per (b, d): relu(pooled^T @ W1 + b1) @ W2 + b2 ----
  if (tid < ND) {
    float p[NH];
#pragma unroll
    for (int hh = 0; hh < NH; ++hh) p[hh] = pooled[hh * ND + tid];
    float tk[16];
#pragma unroll
    for (int k = 0; k < 16; ++k) {
      float a = b1[k];
#pragma unroll
      for (int hh = 0; hh < NH; ++hh) a = fmaf(p[hh], W1[hh * 16 + k], a);
      tk[k] = fmaxf(a, 0.f);
    }
#pragma unroll
    for (int o = 0; o < 8; ++o) {
      float a = b2[o];
#pragma unroll
      for (int k = 0; k < 16; ++k) a = fmaf(tk[k], W2[k * 8 + o], a);
      wordx[(size_t)b * 512 + tid * 8 + o] = a;
    }
  }
}

// ---------------------------------------------------------------------------
// Kernel 4: word gather + segment mean -> name_ft (2048 x 512), concat = d_out
// ---------------------------------------------------------------------------
__global__ void k_seg(const float* __restrict__ wordx,
                      const int*   __restrict__ wcode,
                      const int*   __restrict__ nwords,
                      float*       __restrict__ out) {
  __shared__ int sstart, scnt;
  const int n = blockIdx.x;
  if (threadIdx.x == 0) {
    int st = 0;
    for (int m = 0; m < n; ++m) st += nwords[m];
    sstart = st; scnt = nwords[n];
  }
  __syncthreads();
  const int st = sstart, c = scnt;
  const float inv = 1.f / (float)c;
  for (int colc = threadIdx.x; colc < 512; colc += blockDim.x) {
    float s = 0.f;
    for (int w = 0; w < c; ++w)
      s += wordx[(size_t)wcode[st + w] * 512 + colc];
    out[(size_t)n * 512 + colc] = s * inv;
  }
}

// ---------------------------------------------------------------------------
extern "C" void kernel_launch(void* const* d_in, const int* in_sizes, int n_in,
                              void* d_out, int out_size, void* d_ws, size_t ws_size,
                              hipStream_t stream) {
  (void)in_sizes; (void)n_in; (void)out_size; (void)ws_size;
  const float* emb   = (const float*)d_in[0];
  const float* Wc    = (const float*)d_in[1];
  const float* bc    = (const float*)d_in[2];
  const float* Wp    = (const float*)d_in[3];
  const float* bp    = (const float*)d_in[4];
  const float* gamma = (const float*)d_in[5];
  const float* beta  = (const float*)d_in[6];
  const float* W1    = (const float*)d_in[7];
  const float* b1    = (const float*)d_in[8];
  const float* W2    = (const float*)d_in[9];
  const float* b2    = (const float*)d_in[10];
  const int*   ccode = (const int*)d_in[11];
  const int*   wcode = (const int*)d_in[12];
  const int*   nwrds = (const int*)d_in[13];
  // d_in[14] (n_names) only defines the split points; concatenated output == name_ft

  char* ws = (char*)d_ws;
  _Float16* tabQK = (_Float16*)ws;                               // 4 MB
  _Float16* tabV  = (_Float16*)(ws + 4u * 1024 * 1024);          // 2 MB
  _Float16* posQ  = (_Float16*)(ws + 6u * 1024 * 1024);          // 16 KB each
  _Float16* posK  = posQ + NH * NL * ND;
  _Float16* posV  = posK + NH * NL * ND;
  float*    wordx = (float*)(ws + 6u * 1024 * 1024 + 64u * 1024); // 16 MB

  k_prep_char<<<(NV * 1536 + 255) / 256, 256, 0, stream>>>(emb, Wc, bc, tabQK, tabV);
  k_prep_pos<<<1, 256, 0, stream>>>(Wp, bp, posQ, posK, posV);
  k_attn<<<NB, 128, 0, stream>>>(tabQK, tabV, posQ, posK, posV, ccode,
                                 gamma, beta, W1, b1, W2, b2, wordx);
  k_seg<<<NNAMES, 128, 0, stream>>>(wordx, wcode, nwrds, (float*)d_out);
}
